// center_count_44418551775926
// MI455X (gfx1250) — matrix operand order repaired and checked by reading, not compiled, not run on hardware
//
#include <hip/hip_runtime.h>

// ---------------------------------------------------------------------------
// Per-class mean (segment-sum / count) on MI455X (gfx1250), expressed as a
// one-hot GEMM so the 40-way class scatter runs on the WMMA pipe:
//   D[f, c] = sum_k F[k, f] * OneHot[k, c]      (then divided by counts)
// f32 features are split into bf16 hi + bf16 lo and fed to two
// v_wmma_f32_16x16x32_bf16 per tile (f32 accumulate) to retain ~f32 accuracy.
// ---------------------------------------------------------------------------

typedef __bf16 v16bf __attribute__((ext_vector_type(16)));
typedef float  v8f   __attribute__((ext_vector_type(8)));

#define NUM_CLASSES 40
#define FEAT        3648
#define BATCH       65536
#define CHUNKS      (BATCH / 32)   // 2048 K-chunks of 32 batch rows
#define BSLABS      16             // grid.y
#define WAVES       4              // waves per block (blockDim = 128)
#define OUT_ELEMS   (NUM_CLASSES * FEAT)

__global__ __launch_bounds__(256)
void zero_kernel(float* __restrict__ out, float* __restrict__ counts) {
    int i = blockIdx.x * 256 + threadIdx.x;
    if (i < OUT_ELEMS)    out[i]    = 0.0f;
    if (i < NUM_CLASSES)  counts[i] = 0.0f;
}

__global__ __launch_bounds__(256)
void count_kernel(const int* __restrict__ las, float* __restrict__ counts) {
    __shared__ int bins[NUM_CLASSES];
    int t = threadIdx.x;
    if (t < NUM_CLASSES) bins[t] = 0;
    __syncthreads();
    int i = blockIdx.x * 256 + t;
    int lab = las[i];
    atomicAdd(&bins[lab], 1);
    __syncthreads();
    if (t < NUM_CLASSES) {
        int b = bins[t];
        if (b != 0) atomicAdd(&counts[t], (float)b);
    }
}

__global__ __launch_bounds__(128)
void segsum_wmma_kernel(const float* __restrict__ fts,
                        const int*   __restrict__ las,
                        float*       __restrict__ out) {
    const int lane = threadIdx.x & 31;
    const int wave = threadIdx.x >> 5;
    const int m    = lane & 15;   // A: feature row within tile; B/C: class col
    const int half = lane >> 4;   // lane half selects K sub-range
    const int f0   = blockIdx.x * 16;

    v8f c0 = {}; // classes  0..15
    v8f c1 = {}; // classes 16..31
    v8f c2 = {}; // classes 32..47 (only 32..39 stored)

    const int wstride = BSLABS * WAVES; // 64 batch-waves
    for (int chunk = blockIdx.y * WAVES + wave; chunk < CHUNKS; chunk += wstride) {
        const int k0 = chunk * 32;

        // --- B: one-hot 32(K) x 16(N=classes), bf16. Lane n<16 holds K=0..15,
        //     lane n+16 holds K=16..31, element e <-> K offset e.
        int myLab = las[k0 + lane];
        const int kb = half * 16;
        v16bf b0, b1, b2;
        const __bf16 one  = (__bf16)1.0f;
        const __bf16 zero = (__bf16)0.0f;
#pragma unroll
        for (int e = 0; e < 16; ++e) {
            int lab = __shfl(myLab, kb + e, 32);   // ds_bpermute broadcast
            b0[e] = (lab == m)      ? one : zero;
            b1[e] = (lab == m + 16) ? one : zero;
            b2[e] = (lab == m + 32) ? one : zero;
        }

        // --- A: features 16(M) x 32(K), split into bf16 hi + lo.
        //     Lane half 0: K = e                     -> 0..15
        //     Lane half 1: +16                       -> 16..31
        v16bf ahi, alo;
#pragma unroll
        for (int e = 0; e < 16; ++e) {
            int k = e + half * 16;
            float x  = fts[(size_t)(k0 + k) * FEAT + f0 + m];
            __bf16 h = (__bf16)x;
            ahi[e] = h;
            alo[e] = (__bf16)(x - (float)h);
        }

        // --- 6 WMMAs: (hi + lo) x 3 class tiles, f32 accumulate.
        c0 = __builtin_amdgcn_wmma_f32_16x16x32_bf16(false, ahi, false, b0, (short)0, c0, false, false);
        c0 = __builtin_amdgcn_wmma_f32_16x16x32_bf16(false, alo, false, b0, (short)0, c0, false, false);
        c1 = __builtin_amdgcn_wmma_f32_16x16x32_bf16(false, ahi, false, b1, (short)0, c1, false, false);
        c1 = __builtin_amdgcn_wmma_f32_16x16x32_bf16(false, alo, false, b1, (short)0, c1, false, false);
        c2 = __builtin_amdgcn_wmma_f32_16x16x32_bf16(false, ahi, false, b2, (short)0, c2, false, false);
        c2 = __builtin_amdgcn_wmma_f32_16x16x32_bf16(false, alo, false, b2, (short)0, c2, false, false);
    }

    // C layout: VGPR r, lanes 0-15 -> M=r, lanes 16-31 -> M=r+8; N = lane&15.
    // M = feature offset, N = class within tile.
    const int featb = f0 + half * 8;
#pragma unroll
    for (int r = 0; r < 8; ++r) {
        atomicAdd(&out[(m +  0) * FEAT + featb + r], c0[r]);
        atomicAdd(&out[(m + 16) * FEAT + featb + r], c1[r]);
        if (m + 32 < NUM_CLASSES)
            atomicAdd(&out[(m + 32) * FEAT + featb + r], c2[r]);
    }
}

__global__ __launch_bounds__(256)
void finalize_kernel(float* __restrict__ out, const float* __restrict__ counts) {
    int i = blockIdx.x * 256 + threadIdx.x;
    if (i >= OUT_ELEMS) return;
    int c = i / FEAT;
    float cnt = counts[c];
    float s   = out[i];
    out[i] = (cnt > 0.0f) ? (s / cnt) : 0.0f;
}

extern "C" void kernel_launch(void* const* d_in, const int* in_sizes, int n_in,
                              void* d_out, int out_size, void* d_ws, size_t ws_size,
                              hipStream_t stream) {
    const float* fts = (const float*)d_in[0];
    const int*   las = (const int*)d_in[1];
    float* out    = (float*)d_out;
    float* counts = (float*)d_ws;   // 40 floats of scratch

    (void)in_sizes; (void)n_in; (void)out_size; (void)ws_size;

    // 1) zero accumulators (atomics require clean state every call)
    zero_kernel<<<(OUT_ELEMS + 255) / 256, 256, 0, stream>>>(out, counts);

    // 2) class histogram (LDS-privatized)
    count_kernel<<<BATCH / 256, 256, 0, stream>>>(las, counts);

    // 3) one-hot GEMM segment-sum on the WMMA pipe
    dim3 grid(FEAT / 16, BSLABS);
    segsum_wmma_kernel<<<grid, WAVES * 32, 0, stream>>>(fts, las, out);

    // 4) divide by counts (count==0 -> 0)
    finalize_kernel<<<(OUT_ELEMS + 255) / 256, 256, 0, stream>>>(out, counts);
}
